// MABClean_83167746719797
// MI455X (gfx1250) — matrix operand order repaired and checked by reading, not compiled, not run on hardware
//
#include <hip/hip_runtime.h>
#include <hip/hip_bf16.h>

// ---------------------------------------------------------------------------
// MAB (set-transformer attention block) for MI455X / gfx1250.
// Compute-bound (~22.6 GFLOP vs ~3us of HBM traffic at 23.3 TB/s): all GEMMs
// + attention on v_wmma_f32_16x16x32_bf16 (wave32). Flash softmax processes
// 128 keys per online update (fixed-cost row-reduction butterflies amortize
// to 0.5 ds_bpermute/key); exponentials in native exp2 domain. V-transpose
// stages through LDS with global_load_async_to_lds_b128 (ASYNCcnt path).
// ---------------------------------------------------------------------------

typedef __bf16 bf16_t;
typedef __attribute__((ext_vector_type(16))) __bf16 v16bf;
typedef __attribute__((ext_vector_type(8)))  __bf16 v8bf;
typedef __attribute__((ext_vector_type(8)))  float  v8f;

__device__ __forceinline__ bf16_t f2bf(float f) {
    unsigned u = __builtin_bit_cast(unsigned, f);
    u += 0x7FFFu + ((u >> 16) & 1u);            // round-to-nearest-even
    unsigned short s = (unsigned short)(u >> 16);
    return __builtin_bit_cast(bf16_t, s);
}

// ---------------------------------------------------------------------------
// Per-batch mean / rstd over Nx*D elements (setnorm statistics)
// ---------------------------------------------------------------------------
__global__ void stats_kernel(const float* __restrict__ x, float* __restrict__ mv,
                             int per_batch) {
    __shared__ float ss[256], ss2[256];
    const float* p = x + (size_t)blockIdx.x * per_batch;
    float s = 0.f, s2 = 0.f;
    for (int i = threadIdx.x; i < per_batch; i += blockDim.x) {
        float v = p[i]; s += v; s2 += v * v;
    }
    ss[threadIdx.x] = s; ss2[threadIdx.x] = s2;
    __syncthreads();
    for (int o = 128; o > 0; o >>= 1) {
        if ((int)threadIdx.x < o) {
            ss[threadIdx.x]  += ss[threadIdx.x + o];
            ss2[threadIdx.x] += ss2[threadIdx.x + o];
        }
        __syncthreads();
    }
    if (threadIdx.x == 0) {
        float m   = ss[0] / per_batch;
        float var = ss2[0] / per_batch - m * m;
        mv[2 * blockIdx.x]     = m;
        mv[2 * blockIdx.x + 1] = rsqrtf(fmaxf(var, 0.f) + 1e-5f);
    }
}

// normalize + per-feature affine (+ optional relu), emit bf16
__global__ void norm_affine_bf16(const float* __restrict__ x, const float* __restrict__ mv,
                                 const float* __restrict__ w, const float* __restrict__ bia,
                                 bf16_t* __restrict__ out, int per_batch, int D, int relu) {
    size_t i = (size_t)blockIdx.x * blockDim.x + threadIdx.x;
    int b = (int)(i / (size_t)per_batch);
    int c = (int)(i % (size_t)D);
    float m = mv[2 * b], r = mv[2 * b + 1];
    float v = (x[i] - m) * r * w[c] + bia[c];
    if (relu) v = fmaxf(v, 0.f);
    out[i] = f2bf(v);
}

__global__ void to_bf16_kernel(const float* __restrict__ x, bf16_t* __restrict__ out, int n) {
    int i = blockIdx.x * blockDim.x + threadIdx.x;
    if (i < n) out[i] = f2bf(x[i]);
}

// ---------------------------------------------------------------------------
// GEMM: C[M,N] = A[M,K](bf16) @ W[N,K]^T(bf16) + bias, K % 32 == 0.
// A-frag: lane m=lane&15; elems e<8 -> k = 8*(lane>>4)+e; e>=8 -> 16+8*(lane>>4)+(e-8)
// B-frag: lane n=lane&15; same k order as A-frag (two 8-elem chunks of W row n)
// C-frag: lane n=lane&15; vgpr r    -> m = r + 8*(lane>>4)
// MODE 0: bf16 store; MODE 1: fp32 store + residual add.
// ---------------------------------------------------------------------------
template<int MODE>
__global__ __launch_bounds__(128)
void gemm_bf16(const bf16_t* __restrict__ A, const bf16_t* __restrict__ W,
               const float* __restrict__ bias, const float* __restrict__ resid,
               void* __restrict__ outp, int M, int N, int K) {
    const int wave = threadIdx.x >> 5, lane = threadIdx.x & 31;
    const int hi = lane >> 4, lm = lane & 15;
    const int m0 = (blockIdx.x * 4 + wave) * 16;
    const int n0 = blockIdx.y * 64;

    v8f acc[4] = {};
    const bf16_t* arow = A + (size_t)(m0 + lm) * K;

    for (int k0 = 0; k0 < K; k0 += 32) {
        v8bf alo = *(const v8bf*)(arow + k0 + 8 * hi);
        v8bf ahi = *(const v8bf*)(arow + k0 + 16 + 8 * hi);
        v16bf af;
#pragma unroll
        for (int i = 0; i < 8; ++i) { af[i] = alo[i]; af[i + 8] = ahi[i]; }
#pragma unroll
        for (int t = 0; t < 4; ++t) {
            const bf16_t* wrow = W + (size_t)(n0 + t * 16 + lm) * K + k0;
            v8bf wlo = *(const v8bf*)(wrow + 8 * hi);
            v8bf whi = *(const v8bf*)(wrow + 16 + 8 * hi);
            v16bf wf;
#pragma unroll
            for (int i = 0; i < 8; ++i) { wf[i] = wlo[i]; wf[i + 8] = whi[i]; }
            acc[t] = __builtin_amdgcn_wmma_f32_16x16x32_bf16(
                false, af, false, wf, (short)0, acc[t], false, false);
        }
    }

#pragma unroll
    for (int t = 0; t < 4; ++t) {
        int n = n0 + t * 16 + lm;
        float bv = bias[n];
#pragma unroll
        for (int r = 0; r < 8; ++r) {
            size_t idx = (size_t)(m0 + r + 8 * hi) * N + n;
            float v = acc[t][r] + bv;
            if (MODE == 0) ((bf16_t*)outp)[idx] = f2bf(v);
            else           ((float*)outp)[idx]  = v + resid[idx];
        }
    }
}

// ---------------------------------------------------------------------------
// Transpose bf16 [B,N,D] -> [B,D,N] via async global->LDS staging.
// Each of 128 threads issues one global_load_async_to_lds_b128 (16 B) into a
// 32x40 padded tile (80 B rows keep every async dst 16 B aligned), waits on
// ASYNCcnt + barrier, then gathers 8 strided LDS elems and emits one
// global_store_b128 of the transposed row segment.
// ---------------------------------------------------------------------------
__global__ __launch_bounds__(128)
void transpose_bf16(const bf16_t* __restrict__ in, bf16_t* __restrict__ out,
                    int N, int D) {
    __shared__ __align__(16) bf16_t tile[32][40];
    const bf16_t* src = in + (size_t)blockIdx.z * N * D;
    bf16_t* dst = out + (size_t)blockIdx.z * N * D;
    const int n0 = blockIdx.x * 32, d0 = blockIdx.y * 32;
    const int tid = threadIdx.x;

    // async copy: thread -> row j = tid/4, 8-elem chunk c = tid%4
    {
        int j = tid >> 2, c = tid & 3;
        unsigned lds_off = (unsigned)(uintptr_t)&tile[j][c * 8];
        unsigned long long ga =
            (unsigned long long)(uintptr_t)(src + (size_t)(n0 + j) * D + d0 + c * 8);
        asm volatile("global_load_async_to_lds_b128 %0, %1, off"
                     :: "v"(lds_off), "v"(ga) : "memory");
    }
    asm volatile("s_wait_asynccnt 0" ::: "memory");
    __syncthreads();

    // transposed store: thread -> out row i = tid/4, 8 columns kbase..kbase+7
    {
        int i = tid >> 2, kbase = (tid & 3) * 8;
        v8bf v;
#pragma unroll
        for (int kk = 0; kk < 8; ++kk) v[kk] = tile[kbase + kk][i];
        *(v8bf*)(dst + (size_t)(d0 + i) * N + n0 + kbase) = v;
    }
}

// ---------------------------------------------------------------------------
// Flash attention, 128-key chunks: one wave = 16 query rows of one (b,h).
//  - 8 WMMAs -> S(16x128) fp32 held in registers
//  - one online-softmax update per chunk: row max/sum butterflies (4-stage
//    half-wave shfl_xor) cost is fixed per chunk -> 0.5 ds_bpermute per key
//  - exponentials in native exp2 domain: p = exp2(fma(s, SC, -m))
//  - P transposed C-layout -> A-layout through per-wave LDS (16x128 bf16)
//  - O accumulated fp32 as two 16x16 tiles over head-dim 32
// scale folded: SC = 1/sqrt(256) * log2(e)
// ---------------------------------------------------------------------------
__global__ __launch_bounds__(128)
void attention_kernel(const bf16_t* __restrict__ Q, const bf16_t* __restrict__ Km,
                      const bf16_t* __restrict__ Vt, bf16_t* __restrict__ O) {
    constexpr int D = 256, NX = 2048, NY = 2048, CK = 128;
    constexpr float SC = 0.0625f * 1.4426950408889634f;
    const int wave = threadIdx.x >> 5, lane = threadIdx.x & 31;
    const int hi = lane >> 4, lm = lane & 15;
    const int q0  = (blockIdx.x * 4 + wave) * 16;
    const int hd0 = blockIdx.y * 32;
    const int b   = blockIdx.z;

    const bf16_t* Qb = Q  + (size_t)b * NX * D;
    const bf16_t* Kb = Km + (size_t)b * NY * D;
    const bf16_t* Vb = Vt + ((size_t)b * D + hd0) * NY;

    // Q fragment (A-layout), loaded once
    const bf16_t* qrow = Qb + (size_t)(q0 + lm) * D + hd0;
    v8bf qlo = *(const v8bf*)(qrow + 8 * hi);
    v8bf qhi = *(const v8bf*)(qrow + 16 + 8 * hi);
    v16bf qf;
#pragma unroll
    for (int i = 0; i < 8; ++i) { qf[i] = qlo[i]; qf[i + 8] = qhi[i]; }

    v8f oacc0 = {}, oacc1 = {};
    float rmax[8], rsum[8];
#pragma unroll
    for (int r = 0; r < 8; ++r) { rmax[r] = -3.0e38f; rsum[r] = 0.f; }

    __shared__ __align__(32) bf16_t pback[4][16 * CK];
    bf16_t* pb = pback[wave];

    for (int ky = 0; ky < NY; ky += CK) {
        if (ky + CK < NY) {   // uniform: prefetch next chunk (global_prefetch_b8)
            __builtin_prefetch(Kb + (size_t)(ky + CK + lane) * D + hd0, 0, 1);
            __builtin_prefetch(Vb + (size_t)lane * NY + ky + CK, 0, 1);
        }

        // S = Q K^T for 128 keys: 8 tiles of 16x16
        v8f s[8];
        v8f zero = {};
#pragma unroll
        for (int t = 0; t < 8; ++t) {
            const bf16_t* krow = Kb + (size_t)(ky + t * 16 + lm) * D + hd0;
            v8bf klo = *(const v8bf*)(krow + 8 * hi);
            v8bf khi = *(const v8bf*)(krow + 16 + 8 * hi);
            v16bf kf;
#pragma unroll
            for (int i = 0; i < 8; ++i) { kf[i] = klo[i]; kf[i + 8] = khi[i]; }
            s[t] = __builtin_amdgcn_wmma_f32_16x16x32_bf16(
                false, qf, false, kf, (short)0, zero, false, false);
        }

        // online softmax update (exp2 domain)
#pragma unroll
        for (int r = 0; r < 8; ++r) {
            float mx = s[0][r];
#pragma unroll
            for (int t = 1; t < 8; ++t) mx = fmaxf(mx, s[t][r]);
            mx = fmaxf(mx, __shfl_xor(mx, 1, 32));
            mx = fmaxf(mx, __shfl_xor(mx, 2, 32));
            mx = fmaxf(mx, __shfl_xor(mx, 4, 32));
            mx = fmaxf(mx, __shfl_xor(mx, 8, 32));
            float nm   = fmaxf(rmax[r], mx * SC);
            float corr = __builtin_amdgcn_exp2f(rmax[r] - nm);
            rmax[r] = nm;
            float ps = 0.f;
#pragma unroll
            for (int t = 0; t < 8; ++t) {
                float p = __builtin_amdgcn_exp2f(__builtin_fmaf(s[t][r], SC, -nm));
                ps += p;
                pb[(r + 8 * hi) * CK + t * 16 + lm] = f2bf(p);  // LDS [m][k]
            }
            ps += __shfl_xor(ps, 1, 32);
            ps += __shfl_xor(ps, 2, 32);
            ps += __shfl_xor(ps, 4, 32);
            ps += __shfl_xor(ps, 8, 32);
            rsum[r] = rsum[r] * corr + ps;
            oacc0[r] *= corr;
            oacc1[r] *= corr;
        }
        asm volatile("s_wait_dscnt 0" ::: "memory");   // LDS in-order per wave

        // O += P V over 4 sub-chunks of 32 keys
#pragma unroll
        for (int kk = 0; kk < 4; ++kk) {
            v16bf pf = *(const v16bf*)(pb + lm * CK + kk * 32 + 16 * hi);
            v16bf vf0 = *(const v16bf*)(Vb + (size_t)lm * NY + ky + kk * 32 + 16 * hi);
            v16bf vf1 = *(const v16bf*)(Vb + (size_t)(16 + lm) * NY + ky + kk * 32 + 16 * hi);
            oacc0 = __builtin_amdgcn_wmma_f32_16x16x32_bf16(
                false, pf, false, vf0, (short)0, oacc0, false, false);
            oacc1 = __builtin_amdgcn_wmma_f32_16x16x32_bf16(
                false, pf, false, vf1, (short)0, oacc1, false, false);
        }
    }

    bf16_t* orow = O + (size_t)b * NX * D;
#pragma unroll
    for (int r = 0; r < 8; ++r) {
        int m = q0 + r + 8 * hi;
        float inv = 1.0f / rsum[r];
        orow[(size_t)m * D + hd0 + lm]      = f2bf(oacc0[r] * inv);
        orow[(size_t)m * D + hd0 + 16 + lm] = f2bf(oacc1[r] * inv);
    }
}

// ---------------------------------------------------------------------------
extern "C" void kernel_launch(void* const* d_in, const int* in_sizes, int n_in,
                              void* d_out, int out_size, void* d_ws, size_t ws_size,
                              hipStream_t stream) {
    (void)in_sizes; (void)n_in; (void)out_size; (void)ws_size;
    constexpr int B = 4, NX = 2048, NY = 2048, D = 256;
    constexpr size_t PB  = (size_t)NX * D;   // per-batch elems
    constexpr size_t TOT = (size_t)B * PB;   // 2,097,152

    const float* X    = (const float*)d_in[0];
    const float* Y    = (const float*)d_in[1];
    const float* Wq   = (const float*)d_in[2];
    const float* bq   = (const float*)d_in[3];
    const float* Wk   = (const float*)d_in[4];
    const float* bk   = (const float*)d_in[5];
    const float* Wv   = (const float*)d_in[6];
    const float* bv   = (const float*)d_in[7];
    const float* Wo   = (const float*)d_in[8];
    const float* bo   = (const float*)d_in[9];
    const float* Wres = (const float*)d_in[10];
    const float* bres = (const float*)d_in[11];
    const float* nq_w = (const float*)d_in[12];
    const float* nq_b = (const float*)d_in[13];
    const float* nk_w = (const float*)d_in[14];
    const float* nk_b = (const float*)d_in[15];
    const float* n0_w = (const float*)d_in[16];
    const float* n0_b = (const float*)d_in[17];
    float* out = (float*)d_out;

    char* ws = (char*)d_ws;
    size_t off = 0;
    auto alloc = [&](size_t bytes) {
        size_t o = off; off += (bytes + 255) & ~(size_t)255; return o;
    };
    float*  st_x   = (float*)(ws + alloc(64));
    float*  st_y   = (float*)(ws + alloc(64));
    float*  st_h   = (float*)(ws + alloc(64));
    bf16_t* wq_bf  = (bf16_t*)(ws + alloc((size_t)D * D * 2));
    bf16_t* wk_bf  = (bf16_t*)(ws + alloc((size_t)D * D * 2));
    bf16_t* wv_bf  = (bf16_t*)(ws + alloc((size_t)D * D * 2));
    bf16_t* wo_bf  = (bf16_t*)(ws + alloc((size_t)D * D * 2));
    bf16_t* wr_bf  = (bf16_t*)(ws + alloc((size_t)D * D * 2));
    bf16_t* xn     = (bf16_t*)(ws + alloc(TOT * 2));
    bf16_t* yn     = (bf16_t*)(ws + alloc(TOT * 2));
    bf16_t* ybf    = (bf16_t*)(ws + alloc(TOT * 2));
    bf16_t* qb     = (bf16_t*)(ws + alloc(TOT * 2));
    bf16_t* kb     = (bf16_t*)(ws + alloc(TOT * 2));
    bf16_t* vb     = (bf16_t*)(ws + alloc(TOT * 2));
    bf16_t* vt     = (bf16_t*)(ws + alloc(TOT * 2));
    float*  hbuf   = (float*)(ws + alloc(TOT * 4));
    bf16_t* obf    = xn;    // reuse: Xn dead after Q projection
    bf16_t* hn     = ybf;   // reuse: Y(bf16) dead after V projection

    // setnorm stats + bf16 staging
    stats_kernel<<<B, 256, 0, stream>>>(X, st_x, (int)PB);
    stats_kernel<<<B, 256, 0, stream>>>(Y, st_y, (int)PB);
    norm_affine_bf16<<<(int)(TOT / 256), 256, 0, stream>>>(X, st_x, nq_w, nq_b, xn, (int)PB, D, 0);
    norm_affine_bf16<<<(int)(TOT / 256), 256, 0, stream>>>(Y, st_y, nk_w, nk_b, yn, (int)PB, D, 0);
    to_bf16_kernel<<<(int)(TOT / 256), 256, 0, stream>>>(Y, ybf, (int)TOT);
    to_bf16_kernel<<<(D * D) / 256, 256, 0, stream>>>(Wq, wq_bf, D * D);
    to_bf16_kernel<<<(D * D) / 256, 256, 0, stream>>>(Wk, wk_bf, D * D);
    to_bf16_kernel<<<(D * D) / 256, 256, 0, stream>>>(Wv, wv_bf, D * D);
    to_bf16_kernel<<<(D * D) / 256, 256, 0, stream>>>(Wo, wo_bf, D * D);
    to_bf16_kernel<<<(D * D) / 256, 256, 0, stream>>>(Wres, wr_bf, D * D);

    // Q/K/V projections (flattened over batch)
    dim3 ggrid(B * NX / 64, D / 64);
    gemm_bf16<0><<<ggrid, 128, 0, stream>>>(xn,  wq_bf, bq, nullptr, qb, B * NX, D, D);
    gemm_bf16<0><<<ggrid, 128, 0, stream>>>(yn,  wk_bf, bk, nullptr, kb, B * NY, D, D);
    gemm_bf16<0><<<ggrid, 128, 0, stream>>>(ybf, wv_bf, bv, nullptr, vb, B * NY, D, D);

    // V -> head-dim-major for contiguous B-fragments (async-to-LDS staging)
    transpose_bf16<<<dim3(NY / 32, D / 32, B), 128, 0, stream>>>(vb, vt, NY, D);

    // flash attention: grid (q-tiles/4, heads, batch)
    attention_kernel<<<dim3(NX / 64, 8, B), 128, 0, stream>>>(qb, kb, vt, obf);

    // H = X + O @ Wo^T + bo
    gemm_bf16<1><<<ggrid, 128, 0, stream>>>(obf, wo_bf, bo, X, hbuf, B * NX, D, D);

    // out = H + relu(setnorm(H)) @ Wres^T + bres
    stats_kernel<<<B, 256, 0, stream>>>(hbuf, st_h, (int)PB);
    norm_affine_bf16<<<(int)(TOT / 256), 256, 0, stream>>>(hbuf, st_h, n0_w, n0_b, hn, (int)PB, D, 1);
    gemm_bf16<1><<<ggrid, 128, 0, stream>>>(hn, wr_bf, bres, hbuf, out, B * NX, D, D);
}
